// SelfAttention_81389630259576
// MI455X (gfx1250) — matrix operand order repaired and checked by reading, not compiled, not run on hardware
//
#include <hip/hip_runtime.h>
#include <math.h>

typedef __attribute__((ext_vector_type(16))) _Float16 v16h;
typedef __attribute__((ext_vector_type(8)))  _Float16 v8h;
typedef __attribute__((ext_vector_type(8)))  float    v8f;
typedef __attribute__((ext_vector_type(4)))  unsigned int u32x4;
typedef __attribute__((ext_vector_type(8)))  unsigned int u32x8;

constexpr int B_   = 4;
constexpr int S_   = 4096;
constexpr int DIN  = 1024;
constexpr int DQK  = 64;   // == D_V

// ---------------------------------------------------------------- helpers

static __device__ inline v8f wmma_f16(v16h a, v16h b, v8f c) {
  // D = A(16x32 f16) * B(32x16 f16) + C(16x16 f32)
  return __builtin_amdgcn_wmma_f32_16x16x32_f16(
      /*neg_a=*/false, a, /*neg_b=*/false, b,
      /*c_mod=*/(short)0, c, /*reuse_a=*/false, /*reuse_b=*/false);
}

// A-fragment lane data = two contiguous 8x f16 (16B) chunks
static __device__ inline v16h pack16(const _Float16* p0, const _Float16* p1) {
  v8h a = *(const v8h*)p0;
  v8h b = *(const v8h*)p1;
  return __builtin_shufflevector(a, b, 0,1,2,3,4,5,6,7,8,9,10,11,12,13,14,15);
}

static __device__ inline v8f vzero() {
  v8f z;
  #pragma unroll
  for (int r = 0; r < 8; ++r) z[r] = 0.f;
  return z;
}

// Issue a TDM 2D tile load Global->LDS (CDNA5 ISA/§8).
// dims/strides in elements (data_size = 2 bytes). Tracked on TENSORcnt.
static __device__ inline void tdm_load_2d(unsigned int lds_addr, unsigned long long gaddr,
                                          unsigned int tensor_d0, unsigned int tensor_d1,
                                          unsigned int tile_d0, unsigned int tile_d1,
                                          unsigned int stride0) {
  u32x4 g0;
  g0[0] = 1u;                                      // count=1, user descriptor
  g0[1] = lds_addr;                                // lds_addr [63:32]
  g0[2] = (unsigned int)gaddr;                     // global_addr lo
  g0[3] = ((unsigned int)(gaddr >> 32) & 0x01FFFFFFu) | (2u << 30);  // addr hi | type=2
  u32x8 g1;
  g1[0] = (1u << 16);                              // data_size=1 (2 bytes)
  g1[1] = (tensor_d0 & 0xFFFFu) << 16;             // tensor_dim0 lo16 at [63:48]
  g1[2] = ((tensor_d0 >> 16) & 0xFFFFu) | ((tensor_d1 & 0xFFFFu) << 16);
  g1[3] = ((tensor_d1 >> 16) & 0xFFFFu) | ((tile_d0 & 0xFFFFu) << 16);
  g1[4] = (tile_d1 & 0xFFFFu);                     // tile_dim1, tile_dim2=0
  g1[5] = stride0;                                 // tensor_dim0_stride lo32
  g1[6] = 0u;
  g1[7] = 0u;
  asm volatile("tensor_load_to_lds %0, %1" :: "s"(g0), "s"(g1) : "memory");
}

// ------------------------------------------------- 1) weight transpose+cvt
// Wt[f][d] = (f16) W[d][f]    (W: DIN x DQK row-major)
__global__ void wt_kernel(const float* __restrict__ W, _Float16* __restrict__ Wt) {
  int i = blockIdx.x * blockDim.x + threadIdx.x;   // 65536 threads
  if (i < DIN * DQK) {
    int d = i >> 6;
    int f = i & 63;
    Wt[(size_t)f * DIN + d] = (_Float16)W[i];
  }
}

// ------------------------------------------------- 2) QKV projection (WMMA)
// One wave per (16-row tile, projection). Y = relu(X*W + b).
// Q,K stored row-major f16 [b][s][64]; V stored transposed f16 [b][64][s].
__global__ __launch_bounds__(128)
void qkv_proj_kernel(const float* __restrict__ X,
                     const _Float16* __restrict__ WtQ,
                     const _Float16* __restrict__ WtK,
                     const _Float16* __restrict__ WtV,
                     const float* __restrict__ bq,
                     const float* __restrict__ bk,
                     const float* __restrict__ bv,
                     _Float16* __restrict__ Qh,
                     _Float16* __restrict__ Kh,
                     _Float16* __restrict__ Vt) {
  const int wid  = blockIdx.x * (blockDim.x >> 5) + (threadIdx.x >> 5);
  const int lane = threadIdx.x & 31;
  const int lh   = lane >> 4;      // half-wave
  const int m    = lane & 15;

  const int NT   = B_ * S_ / 16;   // 1024 row tiles
  const int tile = wid % NT;
  const int proj = wid / NT;       // 0=Q 1=K 2=V
  const int b    = tile >> 8;      // 256 tiles per batch
  const int s0   = (tile & 255) << 4;

  const _Float16* Wt   = (proj == 0) ? WtQ : (proj == 1) ? WtK : WtV;
  const float*    bias = (proj == 0) ? bq  : (proj == 1) ? bk  : bv;

  v8f acc[4] = {vzero(), vzero(), vzero(), vzero()};

  const float* xrow = X + (size_t)(b * S_ + s0 + m) * DIN;

  for (int kt = 0; kt < DIN / 32; ++kt) {
    const float* p = xrow + kt * 32;
    // A-fragment: row m, K chunks [8h,8h+8) and [16+8h,16+8h+8)
    float4 u0 = *(const float4*)(p + 8 * lh);
    float4 u1 = *(const float4*)(p + 8 * lh + 4);
    float4 u2 = *(const float4*)(p + 16 + 8 * lh);
    float4 u3 = *(const float4*)(p + 16 + 8 * lh + 4);
    // Preload all four weight B-fragments (one clause) before the WMMA burst.
    v16h bf[4];
    #pragma unroll
    for (int ni = 0; ni < 4; ++ni) {
      const _Float16* wp = Wt + (size_t)(ni * 16 + m) * DIN + kt * 32;
      bf[ni] = pack16(wp + 8 * lh, wp + 16 + 8 * lh);
    }
    v16h a;
    a[0]  = (_Float16)u0.x; a[1]  = (_Float16)u0.y; a[2]  = (_Float16)u0.z; a[3]  = (_Float16)u0.w;
    a[4]  = (_Float16)u1.x; a[5]  = (_Float16)u1.y; a[6]  = (_Float16)u1.z; a[7]  = (_Float16)u1.w;
    a[8]  = (_Float16)u2.x; a[9]  = (_Float16)u2.y; a[10] = (_Float16)u2.z; a[11] = (_Float16)u2.w;
    a[12] = (_Float16)u3.x; a[13] = (_Float16)u3.y; a[14] = (_Float16)u3.z; a[15] = (_Float16)u3.w;

    #pragma unroll
    for (int ni = 0; ni < 4; ++ni) acc[ni] = wmma_f16(a, bf[ni], acc[ni]);
  }

  #pragma unroll
  for (int ni = 0; ni < 4; ++ni) {
    float bb = bias[ni * 16 + m];
    if (proj < 2) {
      _Float16* Out = (proj == 0) ? Qh : Kh;
      #pragma unroll
      for (int r = 0; r < 8; ++r) {           // C: row = s0 + 8h + r, col = 16ni + m
        float y = acc[ni][r] + bb;
        y = y > 0.f ? y : 0.f;
        Out[(size_t)(b * S_ + s0 + 8 * lh + r) * DQK + ni * 16 + m] = (_Float16)y;
      }
    } else {
      v8h pk;
      #pragma unroll
      for (int r = 0; r < 8; ++r) {
        float y = acc[ni][r] + bb;
        y = y > 0.f ? y : 0.f;
        pk[r] = (_Float16)y;
      }
      // Vt[b][f][s]: contiguous over r -> one 16B store
      *(v8h*)(Vt + (size_t)(b * DQK + ni * 16 + m) * S_ + s0 + 8 * lh) = pk;
    }
  }
}

// ------------------------------------------------- 3) fused flash attention
// One wave per 16-query tile. S^T = K*Q^T, online softmax, out^T = V^T*P^T.
// TDM double-buffers K/V^T tiles into per-wave LDS (no barriers needed).
__global__ __launch_bounds__(128)
void attn_kernel(const _Float16* __restrict__ Qh,
                 const _Float16* __restrict__ Kh,
                 const _Float16* __restrict__ Vt,
                 const float* __restrict__ mask,
                 float* __restrict__ out) {
  // 4 waves x 2 buffers x (K tile 32x64 + V^T tile 64x32) f16 = 64 KB
  __shared__ _Float16 smem[4 * 2 * 4096];

  const int wlocal = threadIdx.x >> 5;
  const int wid    = blockIdx.x * 4 + wlocal;
  const int lane   = threadIdx.x & 31;
  const int lh     = lane >> 4;
  const int m      = lane & 15;

  const int b  = wid >> 8;            // 256 query tiles per batch
  const int q0 = (wid & 255) << 4;

  _Float16* sbase = &smem[wlocal * 8192];               // this wave's region
  const unsigned int lds_base = (unsigned int)(size_t)sbase;  // LDS byte address

  // Q^T B-fragments (column = query q0+m), reused for all key tiles
  const _Float16* qrow = Qh + (size_t)(b * S_ + q0 + m) * DQK;
  const v16h qb0 = pack16(qrow + 8 * lh, qrow + 16 + 8 * lh);        // feats  0..31
  const v16h qb1 = pack16(qrow + 32 + 8 * lh, qrow + 48 + 8 * lh);   // feats 32..63

  const _Float16* Kb = Kh + (size_t)b * S_ * DQK;
  const _Float16* Vb = Vt + (size_t)b * DQK * S_;

  // ---- TDM tile stage: K tile (32 rows x 64), V^T tile (64 rows x 32)
  auto issue_tiles = [&](int k0, int buf) {
    const unsigned int ldsK = lds_base + (unsigned int)buf * 8192u;
    const unsigned int ldsV = ldsK + 4096u;
    const unsigned long long gK = (unsigned long long)(size_t)(Kb + (size_t)k0 * DQK);
    const unsigned long long gV = (unsigned long long)(size_t)(Vb + k0);
    tdm_load_2d(ldsK, gK, /*tensor_d0=*/DQK, /*tensor_d1=*/S_,
                /*tile_d0=*/DQK, /*tile_d1=*/32, /*stride0=*/DQK);
    tdm_load_2d(ldsV, gV, /*tensor_d0=*/S_, /*tensor_d1=*/DQK,
                /*tile_d0=*/32, /*tile_d1=*/DQK, /*stride0=*/S_);
  };

  const v8f zero = vzero();
  v8f o[4] = {zero, zero, zero, zero};   // out^T accumulators (64 feats x 16 queries)
  float run_m = -1e30f;
  float run_s = 0.f;

  issue_tiles(0, 0);
  int buf = 0;

  for (int k0 = 0; k0 < S_; k0 += 32) {
    if (k0 + 32 < S_) {
      issue_tiles(k0 + 32, buf ^ 1);              // prefetch next tile via TDM
      __builtin_amdgcn_s_wait_tensorcnt(2);       // current tile's 2 ops done
    } else {
      __builtin_amdgcn_s_wait_tensorcnt(0);
    }
    asm volatile("" ::: "memory");

    const _Float16* sK = sbase + buf * 4096;      // [32 keys][64 feats]
    const _Float16* sV = sK + 2048;               // [64 feats][32 keys]

    // ---- S^T tile = K(32x64) * Q^T(64x16): 4 WMMAs -> c0 (keys 0-15), c1 (16-31)
    const _Float16* krt = sK + m * DQK;           // key row m
    const _Float16* krb = sK + (16 + m) * DQK;    // key row 16+m
    v16h at0 = pack16(krt + 8 * lh,      krt + 16 + 8 * lh);  // feats  0..31
    v16h at1 = pack16(krt + 32 + 8 * lh, krt + 48 + 8 * lh);  // feats 32..63
    v16h ab0 = pack16(krb + 8 * lh,      krb + 16 + 8 * lh);
    v16h ab1 = pack16(krb + 32 + 8 * lh, krb + 48 + 8 * lh);

    v8f c0 = wmma_f16(at0, qb0, zero); c0 = wmma_f16(at1, qb1, c0);
    v8f c1 = wmma_f16(ab0, qb0, zero); c1 = wmma_f16(ab1, qb1, c1);

    // ---- online softmax over keys (lane-local: lane column = query q0+m)
    float tmax = -1e30f;
    #pragma unroll
    for (int r = 0; r < 8; ++r) { tmax = fmaxf(tmax, c0[r]); tmax = fmaxf(tmax, c1[r]); }
    tmax = fmaxf(tmax, __shfl_xor(tmax, 16, 32));
    const float new_m = fmaxf(run_m, tmax);
    const float scale = __expf(run_m - new_m);

    float p0[8], p1[8], lsum = 0.f;
    #pragma unroll
    for (int r = 0; r < 8; ++r) {
      p0[r] = __expf(c0[r] - new_m);
      p1[r] = __expf(c1[r] - new_m);
      lsum += p0[r] + p1[r];
    }
    lsum += __shfl_xor(lsum, 16, 32);
    run_s = run_s * scale + lsum;
    run_m = new_m;
    #pragma unroll
    for (int i = 0; i < 4; ++i) o[i] = o[i] * scale;

    // ---- repack P^T (32 keys x 16 queries) into B-fragment layout
    float x0[8], x1[8];
    #pragma unroll
    for (int r = 0; r < 8; ++r) {
      x0[r] = __shfl_xor(p0[r], 16, 32);
      x1[r] = __shfl_xor(p1[r], 16, 32);
    }
    v16h pb;
    #pragma unroll
    for (int r = 0; r < 8; ++r) {
      pb[r]     = (_Float16)((lh == 0) ? p0[r] : x1[r]);  // keys 16h + r
      pb[8 + r] = (_Float16)((lh == 0) ? x0[r] : p1[r]);  // keys 16h + 8 + r
    }

    // ---- out^T += V^T(64x32) * P^T(32x16): 4 WMMAs
    #pragma unroll
    for (int i = 0; i < 4; ++i) {
      const _Float16* vrow = sV + (i * 16 + m) * 32;      // feat row 16i+m
      v16h va = *(const v16h*)(vrow + 16 * lh);
      o[i] = wmma_f16(va, pb, o[i]);
    }

    buf ^= 1;
  }

  // ---- epilogue: out[b][q][f] = out^T(f,q) / l * mask[b][q]
  const float fac = (1.f / run_s) * mask[b * S_ + q0 + m];
  #pragma unroll
  for (int i = 0; i < 4; ++i) {
    float4 w0, w1;
    w0.x = o[i][0] * fac; w0.y = o[i][1] * fac; w0.z = o[i][2] * fac; w0.w = o[i][3] * fac;
    w1.x = o[i][4] * fac; w1.y = o[i][5] * fac; w1.z = o[i][6] * fac; w1.w = o[i][7] * fac;
    float* op = out + (size_t)(b * S_ + q0 + m) * DQK + i * 16 + 8 * lh;
    *(float4*)(op)     = w0;
    *(float4*)(op + 4) = w1;
  }
}

// ---------------------------------------------------------------- launcher
extern "C" void kernel_launch(void* const* d_in, const int* in_sizes, int n_in,
                              void* d_out, int out_size, void* d_ws, size_t ws_size,
                              hipStream_t stream) {
  const float* X    = (const float*)d_in[0];
  const float* mask = (const float*)d_in[1];
  const float* Wq   = (const float*)d_in[2];
  const float* bq   = (const float*)d_in[3];
  const float* Wk   = (const float*)d_in[4];
  const float* bk   = (const float*)d_in[5];
  const float* Wv   = (const float*)d_in[6];
  const float* bv   = (const float*)d_in[7];
  float* out = (float*)d_out;

  char* ws = (char*)d_ws;
  const size_t qkv_bytes = (size_t)B_ * S_ * DQK * sizeof(_Float16);  // 2 MB
  const size_t wt_bytes  = (size_t)DIN * DQK * sizeof(_Float16);      // 128 KB
  _Float16* Qh  = (_Float16*)(ws);
  _Float16* Kh  = (_Float16*)(ws + qkv_bytes);
  _Float16* Vt  = (_Float16*)(ws + 2 * qkv_bytes);
  _Float16* WtQ = (_Float16*)(ws + 3 * qkv_bytes);
  _Float16* WtK = (_Float16*)(ws + 3 * qkv_bytes + wt_bytes);
  _Float16* WtV = (_Float16*)(ws + 3 * qkv_bytes + 2 * wt_bytes);

  // 1) weight transpose + f16 convert (DIN*DQK = 65536 elements each)
  wt_kernel<<<256, 256, 0, stream>>>(Wq, WtQ);
  wt_kernel<<<256, 256, 0, stream>>>(Wk, WtK);
  wt_kernel<<<256, 256, 0, stream>>>(Wv, WtV);

  // 2) QKV projections: 3 * (B*S/16) = 3072 waves, 4 waves/block
  qkv_proj_kernel<<<768, 128, 0, stream>>>(X, WtQ, WtK, WtV, bq, bk, bv, Qh, Kh, Vt);

  // 3) fused attention: B*S/16 = 1024 waves, 4 waves/block
  attn_kernel<<<256, 128, 0, stream>>>(Qh, Kh, Vt, mask, out);
}
